// GAU_72447508349478
// MI455X (gfx1250) — matrix operand order repaired and checked by reading, not compiled, not run on hardware
//
#include <hip/hip_runtime.h>
#include <hip/hip_bf16.h>
#include <stdint.h>

#define DIMX 512
#define QKD  128
#define HID  1024
#define SEQ  4096
#define NB   4
#define M_TOT (NB * SEQ)   // 16384 token rows
#define NCH  (SEQ / 64)    // 64 j-chunks per batch

typedef __attribute__((ext_vector_type(16))) __bf16 v16bf;
typedef __attribute__((ext_vector_type(8)))  float  v8f;

// --------------------------------------------------------------------------
// CDNA5 async global->LDS path (ASYNCcnt), guarded so compile never fails.
// Probe result (round 2 stderr): builtin exists; param 1 is
//   'int __attribute__((vector_size(16))) addrspace(1)*'  (global int4*)
// so pass typed int4 pointers in AS1 (global) / AS3 (LDS).
// --------------------------------------------------------------------------
#if defined(__has_builtin)
#if __has_builtin(__builtin_amdgcn_global_load_async_to_lds_b128) && \
    __has_builtin(__builtin_amdgcn_s_wait_asynccnt)
#define HAVE_ASYNC_LDS 1
#endif
#endif

typedef int v4i_vs __attribute__((vector_size(16)));
typedef __attribute__((address_space(1))) v4i_vs* as1_v4i;
typedef __attribute__((address_space(3))) v4i_vs* as3_v4i;

static __device__ __forceinline__ void cp16_async(const __bf16* g, __bf16* l) {
#if defined(HAVE_ASYNC_LDS)
    __builtin_amdgcn_global_load_async_to_lds_b128(
        (as1_v4i)(__bf16*)g, (as3_v4i)l, 0, 0);
#else
    *(uint4*)l = *(const uint4*)g;
#endif
}

#if defined(HAVE_ASYNC_LDS)
#define WAIT_ASYNC(n) __builtin_amdgcn_s_wait_asynccnt((short)(n))
#else
#define WAIT_ASYNC(n) do { } while (0)
#endif

static __device__ __forceinline__ float silu_f(float x) {
    return x / (1.0f + __expf(-x));
}

// ---------------------------------------------------------------------------
// WMMA fragment helpers (bf16 16x16x32, wave32). C/D per CDNA5 ISA 7.12.2; A/B use this code's contiguous k order (same in both).
// A (16xK row-major): lane L: m = L&15, half = L>>4.
//   VGPR i in 0..7: K = half*16 + 2i, +1
// B (Kx16, source stored transposed as [n][K] row-major so contraction pairs
//   are contiguous): lane L: n = L&15; VGPR j: K = (L>>4)*16 + 2j, +1
// C/D (16x16 f32): VGPR r, lane L: m = r + (L>>4)*8, n = L&15
// ---------------------------------------------------------------------------
static __device__ __forceinline__ v16bf load_fragA(const __bf16* p, int ld, int lane) {
    union { v16bf v; uint32_t u[8]; } f;
    const uint32_t* row = (const uint32_t*)(p + (size_t)(lane & 15) * ld);
    const int kb2 = (lane >> 4) * 8;
#pragma unroll
    for (int i = 0; i < 4; ++i) f.u[i]     = row[kb2 + i];
#pragma unroll
    for (int i = 0; i < 4; ++i) f.u[4 + i] = row[4 + kb2 + i];
    return f.v;
}

static __device__ __forceinline__ v16bf load_fragB(const __bf16* p, int ld, int lane) {
    union { v16bf v; uint32_t u[8]; } f;
    const uint32_t* col = (const uint32_t*)(p + (size_t)(lane & 15) * ld);
    const int kb2 = (lane >> 4) * 8;
#pragma unroll
    for (int j = 0; j < 8; ++j) f.u[j] = col[kb2 + j];
    return f.v;
}

static __device__ __forceinline__ v8f wmma_bf16(v16bf a, v16bf b, v8f c) {
    return __builtin_amdgcn_wmma_f32_16x16x32_bf16(false, a, false, b,
                                                   (short)0, c, false, false);
}

// ---------------------------------------------------------------------------
// Kernel: cast fp32 [K][N] weight to bf16 transposed [N][K]
// ---------------------------------------------------------------------------
__global__ void cast_transpose(const float* __restrict__ in, __bf16* __restrict__ out,
                               int K, int N) {
    int idx = blockIdx.x * 256 + threadIdx.x;
    if (idx < K * N) {
        int k = idx / N;
        int n = idx - k * N;
        out[(size_t)n * K + k] = (__bf16)in[idx];
    }
}

// ---------------------------------------------------------------------------
// Kernel: LayerNorm over DIMX=512, one block (256 thr) per row, bf16 output
// ---------------------------------------------------------------------------
__global__ void __launch_bounds__(256) ln_kernel(const float* __restrict__ x,
                                                 const float* __restrict__ sc,
                                                 const float* __restrict__ bi,
                                                 __bf16* __restrict__ out) {
    __shared__ float rs[256];
    __shared__ float rq[256];
    const int t = threadIdx.x;
    const size_t row = blockIdx.x;
    const float* xr = x + row * DIMX;
    const float a0 = xr[t];
    const float a1 = xr[t + 256];
    rs[t] = a0 + a1;
    rq[t] = a0 * a0 + a1 * a1;
    __syncthreads();
    for (int off = 128; off > 0; off >>= 1) {
        if (t < off) { rs[t] += rs[t + off]; rq[t] += rq[t + off]; }
        __syncthreads();
    }
    const float mu   = rs[0] * (1.0f / DIMX);
    const float var  = rq[0] * (1.0f / DIMX) - mu * mu;
    const float rstd = rsqrtf(var + 1e-5f);
    __bf16* orow = out + row * DIMX;
    orow[t]       = (__bf16)((a0 - mu) * rstd * sc[t]       + bi[t]);
    orow[t + 256] = (__bf16)((a1 - mu) * rstd * sc[t + 256] + bi[t + 256]);
}

// ---------------------------------------------------------------------------
// Kernel: hidden = silu(normed @ W_hidden + b).
// First half  -> vT (bf16, TRANSPOSED [n][token]) so attention can stage it
//                with contiguous async b128 copies.
// Second half -> gate (f32, row-major).
// M=16384, N=2048, K=512. Block = 8 waves; wave computes 16x64.
// grid = (M/16, N/512)
// ---------------------------------------------------------------------------
__global__ void __launch_bounds__(256) hidden_gemm(const __bf16* __restrict__ A,
                                                   const __bf16* __restrict__ Wt,
                                                   const float* __restrict__ bias,
                                                   __bf16* __restrict__ vT,
                                                   float* __restrict__ gout) {
    const int t = threadIdx.x, lane = t & 31, w = t >> 5;
    const int m0 = blockIdx.x * 16;
    const int n0 = blockIdx.y * 512 + w * 64;
    v8f acc[4];
#pragma unroll
    for (int i = 0; i < 4; ++i) acc[i] = v8f{};

    for (int kk = 0; kk < DIMX; kk += 32) {
        v16bf a = load_fragA(A + (size_t)m0 * DIMX + kk, DIMX, lane);
#pragma unroll
        for (int tn = 0; tn < 4; ++tn) {
            v16bf b = load_fragB(Wt + (size_t)(n0 + 16 * tn) * DIMX + kk, DIMX, lane);
            acc[tn] = wmma_bf16(a, b, acc[tn]);
        }
    }
#pragma unroll
    for (int tn = 0; tn < 4; ++tn) {
#pragma unroll
        for (int r = 0; r < 8; ++r) {
            const int m = m0 + r + (lane >> 4) * 8;
            const int n = n0 + 16 * tn + (lane & 15);
            float h = silu_f(acc[tn][r] + bias[n]);
            if (n < HID) vT[(size_t)n * M_TOT + m] = (__bf16)h;
            else         gout[(size_t)m * HID + (n - HID)] = h;
        }
    }
}

// ---------------------------------------------------------------------------
// Kernel: Z = silu(normed @ W_qk + b); q=Z*g0+b0, k=Z*g1+b1 (bf16)
// M=16384, N=128, K=512. 8 waves: 4 m-tiles x 2 n-halves. grid = M/64
// ---------------------------------------------------------------------------
__global__ void __launch_bounds__(256) qk_gemm(const __bf16* __restrict__ A,
                                               const __bf16* __restrict__ Wt,
                                               const float* __restrict__ bias,
                                               const float* __restrict__ gamma,
                                               const float* __restrict__ beta,
                                               __bf16* __restrict__ qout,
                                               __bf16* __restrict__ kout) {
    const int t = threadIdx.x, lane = t & 31, w = t >> 5;
    const int m0 = blockIdx.x * 64 + (w >> 1) * 16;
    const int n0 = (w & 1) * 64;
    v8f acc[4];
#pragma unroll
    for (int i = 0; i < 4; ++i) acc[i] = v8f{};

    for (int kk = 0; kk < DIMX; kk += 32) {
        v16bf a = load_fragA(A + (size_t)m0 * DIMX + kk, DIMX, lane);
#pragma unroll
        for (int tn = 0; tn < 4; ++tn) {
            v16bf b = load_fragB(Wt + (size_t)(n0 + 16 * tn) * DIMX + kk, DIMX, lane);
            acc[tn] = wmma_bf16(a, b, acc[tn]);
        }
    }
#pragma unroll
    for (int tn = 0; tn < 4; ++tn) {
#pragma unroll
        for (int r = 0; r < 8; ++r) {
            const int m = m0 + r + (lane >> 4) * 8;
            const int n = n0 + 16 * tn + (lane & 15);
            float z = silu_f(acc[tn][r] + bias[n]);
            qout[(size_t)m * QKD + n] = (__bf16)(z * gamma[n]       + beta[n]);
            kout[(size_t)m * QKD + n] = (__bf16)(z * gamma[QKD + n] + beta[QKD + n]);
        }
    }
}

// ---------------------------------------------------------------------------
// Attention staging helpers: 16B segments, all contiguous & 16B aligned.
// stage_tile64x128: 64 rows x 128 cols contiguous (q tile / k chunk): 4 ops/thr
// stage_vT: 256 rows (n) x 64 cols (j) from vT[n][token]:             8 ops/thr
// ---------------------------------------------------------------------------
static __device__ __forceinline__ void stage_tile64x128(const __bf16* src,
                                                        __bf16* dst, int t) {
#pragma unroll
    for (int i = 0; i < 4; ++i)
        cp16_async(src + t * 8 + i * 2048, dst + t * 8 + i * 2048);
}

static __device__ __forceinline__ void stage_vT(const __bf16* vT, __bf16* dst,
                                                int bb, int j0, int n0, int t) {
    const __bf16* row = vT + (size_t)(n0 + t) * M_TOT + (size_t)bb * SEQ + j0;
#pragma unroll
    for (int s = 0; s < 8; ++s)
        cp16_async(row + s * 8, dst + t * 64 + s * 8);
}

// ---------------------------------------------------------------------------
// Fused relu^2-attention:
//   out[i,n] = (sum_j relu(q_i.k_j / SEQ)^2 * v[j,n]) * gate[i,n]
// Block (256 thr) owns 64 i-rows x 256 n-cols; streams j in 64-chunks with
// double-buffered async global->LDS staging (12 async ops per thread/chunk).
// grid = (SEQ/64, HID/256, NB). LDS = 16+32+64+8 = 120 KB.
// ---------------------------------------------------------------------------
__global__ void __launch_bounds__(256) attn_kernel(const __bf16* __restrict__ q,
                                                   const __bf16* __restrict__ kmat,
                                                   const __bf16* __restrict__ vT,
                                                   const float* __restrict__ gate,
                                                   __bf16* __restrict__ Vg) {
    __shared__ __bf16 qs[64 * QKD];        // 16 KB q tile
    __shared__ __bf16 ks[2][64 * QKD];     // 2 x 16 KB k chunks
    __shared__ __bf16 vts[2][256 * 64];    // 2 x 32 KB v^T chunks [n][j]
    __shared__ __bf16 Ss[64 * 64];         //  8 KB relu^2 scores (bf16)

    const int t = threadIdx.x, lane = t & 31, w = t >> 5;
    const int i0 = blockIdx.x * 64;
    const int n0 = blockIdx.y * 256;
    const int bb = blockIdx.z;
    const float invS = 1.0f / (float)SEQ;
    const int mt = w & 3;            // local 16-row i-tile of this wave
    const int nb = (w >> 2) * 128;   // local n offset (0 or 128)

    // prologue: q tile (4 ops) + chunk 0 (12 ops)
    stage_tile64x128(q + (size_t)(bb * SEQ + i0) * QKD, qs, t);
    stage_tile64x128(kmat + (size_t)(bb * SEQ) * QKD, ks[0], t);
    stage_vT(vT, vts[0], bb, 0, n0, t);

    v8f acc[8];
#pragma unroll
    for (int i = 0; i < 8; ++i) acc[i] = v8f{};

    for (int jc = 0; jc < NCH; ++jc) {
        const int cur = jc & 1;
        // prefetch next chunk into the other buffer, then wait for current
        if (jc + 1 < NCH) {
            stage_tile64x128(kmat + (size_t)(bb * SEQ + (jc + 1) * 64) * QKD,
                             ks[cur ^ 1], t);
            stage_vT(vT, vts[cur ^ 1], bb, (jc + 1) * 64, n0, t);
            WAIT_ASYNC(12);          // <=12 outstanding -> current chunk landed
        } else {
            WAIT_ASYNC(0);
        }
        __syncthreads();

        // S = relu(q k^T / SEQ)^2 : 16 tiles of 16x16 over 8 waves (2 each)
#pragma unroll
        for (int tt = 0; tt < 2; ++tt) {
            const int tile = w * 2 + tt;
            const int si = tile >> 2, sj = tile & 3;
            v8f s = v8f{};
#pragma unroll
            for (int kk = 0; kk < QKD; kk += 32) {
                v16bf a = load_fragA(qs + si * 16 * QKD + kk, QKD, lane);
                v16bf b = load_fragB(ks[cur] + sj * 16 * QKD + kk, QKD, lane);
                s = wmma_bf16(a, b, s);
            }
#pragma unroll
            for (int r = 0; r < 8; ++r) {
                float v0 = s[r] * invS;
                v0 = (v0 > 0.0f) ? v0 * v0 : 0.0f;
                Ss[(si * 16 + r + (lane >> 4) * 8) * 64 + sj * 16 + (lane & 15)] =
                    (__bf16)v0;
            }
        }
        __syncthreads();

        // acc += S @ v  (K = 64 over the j chunk)
#pragma unroll
        for (int kk = 0; kk < 64; kk += 32) {
            v16bf a = load_fragA(Ss + mt * 16 * 64 + kk, 64, lane);
#pragma unroll
            for (int tn = 0; tn < 8; ++tn) {
                v16bf b = load_fragB(vts[cur] + (size_t)(nb + 16 * tn) * 64 + kk,
                                     64, lane);
                acc[tn] = wmma_bf16(a, b, acc[tn]);
            }
        }
        __syncthreads();   // buffers free for reuse by next prefetch
    }

    // epilogue: multiply by gate, write bf16 for the output GEMM
#pragma unroll
    for (int tn = 0; tn < 8; ++tn) {
#pragma unroll
        for (int r = 0; r < 8; ++r) {
            const int ml = mt * 16 + r + (lane >> 4) * 8;
            const size_t gm = (size_t)(bb * SEQ + i0 + ml);
            const int n = n0 + nb + 16 * tn + (lane & 15);
            float val = acc[tn][r] * gate[gm * HID + n];
            Vg[gm * HID + n] = (__bf16)val;
        }
    }
}

// ---------------------------------------------------------------------------
// Kernel: out = Vg @ W_out + b_out + x   (fp32 out)
// M=16384, N=512, K=1024. Wave computes 16x64. grid = M/16
// ---------------------------------------------------------------------------
__global__ void __launch_bounds__(256) out_gemm(const __bf16* __restrict__ Vg,
                                                const __bf16* __restrict__ Wt,
                                                const float* __restrict__ bias,
                                                const float* __restrict__ x,
                                                float* __restrict__ out) {
    const int t = threadIdx.x, lane = t & 31, w = t >> 5;
    const int m0 = blockIdx.x * 16;
    const int n0 = w * 64;
    v8f acc[4];
#pragma unroll
    for (int i = 0; i < 4; ++i) acc[i] = v8f{};

    for (int kk = 0; kk < HID; kk += 32) {
        v16bf a = load_fragA(Vg + (size_t)m0 * HID + kk, HID, lane);
#pragma unroll
        for (int tn = 0; tn < 4; ++tn) {
            v16bf b = load_fragB(Wt + (size_t)(n0 + 16 * tn) * HID + kk, HID, lane);
            acc[tn] = wmma_bf16(a, b, acc[tn]);
        }
    }
#pragma unroll
    for (int tn = 0; tn < 4; ++tn) {
#pragma unroll
        for (int r = 0; r < 8; ++r) {
            const int m = m0 + r + (lane >> 4) * 8;
            const int n = n0 + 16 * tn + (lane & 15);
            out[(size_t)m * DIMX + n] = acc[tn][r] + bias[n] + x[(size_t)m * DIMX + n];
        }
    }
}

// ---------------------------------------------------------------------------
extern "C" void kernel_launch(void* const* d_in, const int* in_sizes, int n_in,
                              void* d_out, int out_size, void* d_ws, size_t ws_size,
                              hipStream_t stream) {
    (void)in_sizes; (void)n_in; (void)out_size; (void)ws_size;
    const float* x     = (const float*)d_in[0];
    const float* nsc   = (const float*)d_in[1];
    const float* nbi   = (const float*)d_in[2];
    const float* Wh    = (const float*)d_in[3];
    const float* bh    = (const float*)d_in[4];
    const float* Wqk   = (const float*)d_in[5];
    const float* bqk   = (const float*)d_in[6];
    const float* gamma = (const float*)d_in[7];
    const float* beta  = (const float*)d_in[8];
    const float* Wo    = (const float*)d_in[9];
    const float* bo    = (const float*)d_in[10];
    float* out = (float*)d_out;

    char* p = (char*)d_ws;
    auto alloc = [&](size_t bytes) -> char* {
        char* r = p;
        p += (bytes + 255) & ~(size_t)255;
        return r;
    };
    __bf16* WhT    = (__bf16*)alloc((size_t)2 * HID * DIMX * 2);  //  2 MB
    __bf16* WqkT   = (__bf16*)alloc((size_t)QKD * DIMX * 2);      // 128 KB
    __bf16* WoT    = (__bf16*)alloc((size_t)DIMX * HID * 2);      //  1 MB
    __bf16* normed = (__bf16*)alloc((size_t)M_TOT * DIMX * 2);    // 16 MB
    __bf16* vTbuf  = (__bf16*)alloc((size_t)HID * M_TOT * 2);     // 32 MB
    float*  gbuf   = (float*) alloc((size_t)M_TOT * HID * 4);     // 64 MB
    __bf16* qbuf   = (__bf16*)alloc((size_t)M_TOT * QKD * 2);     //  4 MB
    __bf16* kbuf   = (__bf16*)alloc((size_t)M_TOT * QKD * 2);     //  4 MB
    __bf16* Vg     = (__bf16*)alloc((size_t)M_TOT * HID * 2);     // 32 MB

    cast_transpose<<<(DIMX * 2 * HID + 255) / 256, 256, 0, stream>>>(Wh,  WhT,  DIMX, 2 * HID);
    cast_transpose<<<(DIMX * QKD + 255) / 256,     256, 0, stream>>>(Wqk, WqkT, DIMX, QKD);
    cast_transpose<<<(HID * DIMX + 255) / 256,     256, 0, stream>>>(Wo,  WoT,  HID, DIMX);

    ln_kernel<<<M_TOT, 256, 0, stream>>>(x, nsc, nbi, normed);

    hidden_gemm<<<dim3(M_TOT / 16, (2 * HID) / 512), 256, 0, stream>>>(
        normed, WhT, bh, vTbuf, gbuf);
    qk_gemm<<<M_TOT / 64, 256, 0, stream>>>(normed, WqkT, bqk, gamma, beta, qbuf, kbuf);

    attn_kernel<<<dim3(SEQ / 64, HID / 256, NB), 256, 0, stream>>>(
        qbuf, kbuf, vTbuf, gbuf, Vg);

    out_gemm<<<M_TOT / 16, 256, 0, stream>>>(Vg, WoT, bo, x, out);
}
